// DecoderBlock_69002944577941
// MI455X (gfx1250) — matrix operand order repaired and checked by reading, not compiled, not run on hardware
//
#include <hip/hip_runtime.h>
#include <hip/hip_bf16.h>
#include <stdint.h>

#define B_SZ   2
#define L_SZ   2048
#define D_SZ   1024
#define H_SZ   16
#define DFF_SZ 4096
#define DK_SZ  64          // D / H

typedef __bf16 bf16_t;
typedef bf16_t bf16x16 __attribute__((ext_vector_type(16)));
typedef float  f32x8   __attribute__((ext_vector_type(8)));
typedef uint32_t u32x4v __attribute__((ext_vector_type(4)));
typedef int      i32x8v __attribute__((ext_vector_type(8)));
typedef int      i32x4v __attribute__((ext_vector_type(4)));

union Frag16 { bf16x16 v; uint32_t u[8]; };

__device__ __forceinline__ uint16_t f2bf(float f) {
  union { float f; uint32_t u; } c; c.f = f;
  uint32_t u = c.u;
  uint32_t r = u + 0x7FFFu + ((u >> 16) & 1u);   // round-to-nearest-even
  return (uint16_t)(r >> 16);
}

__device__ __forceinline__ float half16_max(float v) {
  v = fmaxf(v, __shfl_xor(v, 1, 32));
  v = fmaxf(v, __shfl_xor(v, 2, 32));
  v = fmaxf(v, __shfl_xor(v, 4, 32));
  v = fmaxf(v, __shfl_xor(v, 8, 32));
  return v;
}
__device__ __forceinline__ float half16_sum(float v) {
  v += __shfl_xor(v, 1, 32);
  v += __shfl_xor(v, 2, 32);
  v += __shfl_xor(v, 4, 32);
  v += __shfl_xor(v, 8, 32);
  return v;
}

// fp32 -> bf16(u16) elementwise
__global__ __launch_bounds__(256)
void cast_f32_bf16(const float* __restrict__ src, uint16_t* __restrict__ dst, int n) {
  int i = blockIdx.x * 256 + threadIdx.x;
  if (i < n) dst[i] = f2bf(src[i]);
}

// ---------------------------------------------------------------------------
// GEMM: C[M,N] = act( A[M,K](bf16) @ W[K,N](bf16) + bias[N] (+ res[M,N]) )
// Workgroup tile 256x64 (8 waves x 32 rows), K-step 32.
// A tile staged by the Tensor Data Mover (wave 0 issues TENSOR_LOAD_TO_LDS);
// W tile staged transposed by all threads. 8 wmma / wave / K-step.
// ---------------------------------------------------------------------------
template<bool RELU, bool HAS_RES, bool OUT_F32, bool OUT_BF16>
__global__ __launch_bounds__(256)
void gemm_bf16(const uint16_t* __restrict__ A,
               const uint16_t* __restrict__ W,
               const float*    __restrict__ bias,
               const float*    __restrict__ res,
               float*          __restrict__ Cf,
               uint16_t*       __restrict__ Cb,
               int M, int N, int K)
{
  __shared__ uint16_t ldsA[256 * 32];     // A tile row-major [256][32]  (16KB)
  __shared__ uint16_t ldsBt[64 * 32];     // W tile transposed [n][k]    (4KB)

  const int tid  = threadIdx.x;
  const int wave = tid >> 5;
  const int lane = tid & 31;
  const int hi   = lane >> 4;
  const int ln   = lane & 15;
  const int m0   = blockIdx.y * 256;
  const int n0   = blockIdx.x * 64;

  const uint32_t ldsA_off = (uint32_t)(uintptr_t)(&ldsA[0]);   // LDS byte offset

  f32x8 acc[2][4] = {};

  for (int k0 = 0; k0 < K; k0 += 32) {
    __syncthreads();
    if (wave == 0) {
      // --- TDM descriptor: 2D tile 32(K) x 256(rows), row stride K elements ---
      const uint64_t ga = (uint64_t)(uintptr_t)(A + (size_t)m0 * K + k0);
      u32x4v g0; i32x8v g1; i32x4v g2 = {}; i32x4v g3 = {}; i32x8v g4 = {};
      g0[0] = 1u;                                           // count=1, user mode
      g0[1] = ldsA_off;                                     // lds_addr
      g0[2] = (uint32_t)ga;                                 // global_addr[31:0]
      g0[3] = (uint32_t)((ga >> 32) & 0x01FFFFFFu) | 0x80000000u; // [56:32] | type=2
      const uint32_t dim0 = (uint32_t)K;                    // tensor dim0 (elements)
      const uint32_t dim1 = (uint32_t)M;                    // tensor dim1 (rows)
      const uint32_t str0 = (uint32_t)K;                    // dim0 stride (elements)
      g1[0] = 0x00010000;                                   // wg_mask=0, data_size=2B
      g1[1] = (int)((dim0 & 0xFFFFu) << 16);                // tensor_dim0[15:0]
      g1[2] = (int)((dim0 >> 16) | ((dim1 & 0xFFFFu) << 16));
      g1[3] = (int)((dim1 >> 16) | (32u << 16));            // tile_dim0 = 32
      g1[4] = (int)256u;                                    // tile_dim1 = 256, tile_dim2 = 0
      g1[5] = (int)str0;                                    // tensor_dim0_stride[31:0]
      g1[6] = 0;                                            // stride[47:32]=0, dim1_stride lo=0
      g1[7] = 0;
      __builtin_amdgcn_tensor_load_to_lds(g0, g1, g2, g3, g4, 0);
    }
    { // stage W 32x64 transposed: thread -> k row tid/8, 8 cols
      const int kr = tid >> 3, nc = (tid & 7) * 8;
      const uint16_t* src = W + (size_t)(k0 + kr) * N + n0 + nc;
      uint4 x = *(const uint4*)src;
      const uint16_t* xs = (const uint16_t*)&x;
#pragma unroll
      for (int j = 0; j < 8; ++j) ldsBt[(nc + j) * 32 + kr] = xs[j];
      if (k0 + 32 < K) __builtin_prefetch((const void*)(src + (size_t)32 * N), 0, 3);
    }
    if (wave == 0) __builtin_amdgcn_s_wait_tensorcnt(0);    // TDM done before barrier
    __syncthreads();

    // Two A fragments for this wave's 32 rows (contiguous k = 16*hi + e, same order as B)
    Frag16 a0, a1;
    {
      const uint16_t* b0 = &ldsA[(wave * 32 + ln) * 32];
      const uint16_t* b1 = &ldsA[(wave * 32 + 16 + ln) * 32];
#pragma unroll
      for (int p = 0; p < 8; ++p) {
        const int kk = hi * 16 + 2 * p;
        a0.u[p] = *(const uint32_t*)(b0 + kk);
        a1.u[p] = *(const uint32_t*)(b1 + kk);
      }
    }
#pragma unroll
    for (int s = 0; s < 4; ++s) {
      Frag16 bfrag;  // B 32x16: N = ln, K = 16*hi + e  -> contiguous in ldsBt row
      const uint16_t* base = &ldsBt[(s * 16 + ln) * 32 + hi * 16];
#pragma unroll
      for (int p = 0; p < 8; ++p) bfrag.u[p] = *(const uint32_t*)(base + 2 * p);
      acc[0][s] = __builtin_amdgcn_wmma_f32_16x16x32_bf16(
          false, a0.v, false, bfrag.v, (short)0, acc[0][s], false, false);
      acc[1][s] = __builtin_amdgcn_wmma_f32_16x16x32_bf16(
          false, a1.v, false, bfrag.v, (short)0, acc[1][s], false, false);
    }
  }

  // epilogue: element r of acc[g][s] is (M = wave*32 + g*16 + r + 8*hi, N = s*16 + ln)
#pragma unroll
  for (int s = 0; s < 4; ++s) {
    const int gn = n0 + s * 16 + ln;
    const float bv = bias[gn];
#pragma unroll
    for (int g = 0; g < 2; ++g)
#pragma unroll
      for (int r = 0; r < 8; ++r) {
        const int gm = m0 + wave * 32 + g * 16 + r + 8 * hi;
        float v = acc[g][s][r] + bv;
        if (HAS_RES) v += res[(size_t)gm * N + gn];
        if (RELU)    v = fmaxf(v, 0.f);
        if (OUT_F32) Cf[(size_t)gm * N + gn] = v;
        if (OUT_BF16) Cb[(size_t)gm * N + gn] = f2bf(v);
      }
  }
}

// ---------------------------------------------------------------------------
// Flash attention, dk = 64, bf16 Q/K/V laid out as [B, L, D] with head h in
// cols [h*64, h*64+64). Each wave owns 16 q rows; kv tiles of 32 keys.
// QK^T: 4 wmma / tile, P@V: 4 wmma / tile. Online softmax via half-wave shfl.
// ---------------------------------------------------------------------------
template<bool CAUSAL>
__global__ __launch_bounds__(256)
void flash_attn_bf16(const uint16_t* __restrict__ Q,
                     const uint16_t* __restrict__ Km,
                     const uint16_t* __restrict__ V,
                     uint16_t* __restrict__ O)
{
  __shared__ uint16_t ldsVt[8 * DK_SZ * 32];  // per-wave V^T tile [dk=64][key=32]
  __shared__ uint16_t ldsP [8 * 16 * 32];     // per-wave P tile [16 q][32 key]

  const int tid  = threadIdx.x;
  const int wave = tid >> 5;
  const int lane = tid & 31;
  const int hi   = lane >> 4;
  const int ln   = lane & 15;
  const int b    = blockIdx.z;
  const int h    = blockIdx.y;
  const int q0   = blockIdx.x * 128 + wave * 16;
  const size_t base = (size_t)b * L_SZ * D_SZ + (size_t)h * DK_SZ;
  const uint16_t* Qb = Q  + base;
  const uint16_t* Kb = Km + base;
  const uint16_t* Vb = V  + base;
  uint16_t* vt = &ldsVt[wave * DK_SZ * 32];
  uint16_t* pp = &ldsP [wave * 16 * 32];

  // Q fragments (16 x 64 -> two 16x32 A-fragments), kept in registers
  Frag16 qf[2];
  {
    const uint16_t* qr = Qb + (size_t)(q0 + ln) * D_SZ;
#pragma unroll
    for (int c = 0; c < 2; ++c)
#pragma unroll
      for (int p = 0; p < 8; ++p) {
        const int kk = c * 32 + hi * 16 + 2 * p;
        qf[c].u[p] = *(const uint32_t*)(qr + kk);
      }
  }

  f32x8 o[4] = {};
  float mrow[8], lrow[8];
#pragma unroll
  for (int r = 0; r < 8; ++r) { mrow[r] = -3.0e38f; lrow[r] = 0.f; }

  const int kend = CAUSAL ? (q0 + 16) : L_SZ;
  for (int kb = 0; kb < kend; kb += 32) {
    { // stage V^T: lane -> one key row, 64 dk values
      const uint16_t* vr = Vb + (size_t)(kb + lane) * D_SZ;
#pragma unroll
      for (int c = 0; c < 8; ++c) {
        uint4 x = *(const uint4*)(vr + c * 8);
        const uint16_t* xs = (const uint16_t*)&x;
#pragma unroll
        for (int j = 0; j < 8; ++j) vt[(c * 8 + j) * 32 + lane] = xs[j];
      }
    }
    // S = (Q K^T): two 16x16 score tiles, contraction over dk=64
    f32x8 s0 = {}, s1 = {};
#pragma unroll
    for (int c = 0; c < 2; ++c) {
      Frag16 kf0, kf1;   // B frag: N = key = ln, K = dk = c*32 + 16*hi + e
      const uint16_t* k0p = Kb + (size_t)(kb + ln)      * D_SZ + c * 32 + hi * 16;
      const uint16_t* k1p = Kb + (size_t)(kb + 16 + ln) * D_SZ + c * 32 + hi * 16;
#pragma unroll
      for (int p = 0; p < 8; ++p) {
        kf0.u[p] = *(const uint32_t*)(k0p + 2 * p);
        kf1.u[p] = *(const uint32_t*)(k1p + 2 * p);
      }
      s0 = __builtin_amdgcn_wmma_f32_16x16x32_bf16(false, qf[c].v, false, kf0.v, (short)0, s0, false, false);
      s1 = __builtin_amdgcn_wmma_f32_16x16x32_bf16(false, qf[c].v, false, kf1.v, (short)0, s1, false, false);
    }
    // online softmax (row M = r + 8*hi lives across the 16 lanes of this half)
    float alpha[8];
#pragma unroll
    for (int r = 0; r < 8; ++r) {
      const int qrow = q0 + r + 8 * hi;
      float a0 = s0[r] * 0.125f;          // 1/sqrt(64)
      float a1 = s1[r] * 0.125f;
      if (CAUSAL) {
        if (kb + ln > qrow)      a0 = -3.0e38f;
        if (kb + 16 + ln > qrow) a1 = -3.0e38f;
      }
      const float t  = half16_max(fmaxf(a0, a1));
      const float mn = fmaxf(mrow[r], t);
      const float al = __expf(mrow[r] - mn);
      const float p0 = __expf(a0 - mn);
      const float p1 = __expf(a1 - mn);
      lrow[r] = lrow[r] * al + half16_sum(p0 + p1);
      mrow[r] = mn;
      alpha[r] = al;
      const int prow = r + 8 * hi;
      pp[prow * 32 + ln]      = f2bf(p0);
      pp[prow * 32 + 16 + ln] = f2bf(p1);
    }
#pragma unroll
    for (int seg = 0; seg < 4; ++seg)
#pragma unroll
      for (int r = 0; r < 8; ++r) o[seg][r] *= alpha[r];

    // P as A-fragment (16 x 32)
    Frag16 pf;
    {
      const uint16_t* pr = pp + ln * 32;
#pragma unroll
      for (int p = 0; p < 8; ++p) {
        const int kk = hi * 16 + 2 * p;
        pf.u[p] = *(const uint32_t*)(pr + kk);
      }
    }
    // O += P @ V  (per dk-segment of 16)
#pragma unroll
    for (int seg = 0; seg < 4; ++seg) {
      Frag16 vf;   // B frag: N = dk = seg*16 + ln, K = key = 16*hi + e
      const uint16_t* vp = &vt[(seg * 16 + ln) * 32 + hi * 16];
#pragma unroll
      for (int p = 0; p < 8; ++p) vf.u[p] = *(const uint32_t*)(vp + 2 * p);
      o[seg] = __builtin_amdgcn_wmma_f32_16x16x32_bf16(false, pf.v, false, vf.v, (short)0, o[seg], false, false);
    }
  }

  // normalize and emit bf16
#pragma unroll
  for (int seg = 0; seg < 4; ++seg)
#pragma unroll
    for (int r = 0; r < 8; ++r) {
      const int qrow = q0 + r + 8 * hi;
      const float vvl = o[seg][r] / lrow[r];
      O[base + (size_t)qrow * D_SZ + seg * 16 + ln] = f2bf(vvl);
    }
}

// ---------------------------------------------------------------------------
extern "C" void kernel_launch(void* const* d_in, const int* in_sizes, int n_in,
                              void* d_out, int out_size, void* d_ws, size_t ws_size,
                              hipStream_t stream) {
  (void)in_sizes; (void)n_in; (void)out_size; (void)ws_size;
  const float* tgt = (const float*)d_in[0];
  const float* enc = (const float*)d_in[1];
  const float* sWq = (const float*)d_in[4];   const float* sbq = (const float*)d_in[5];
  const float* sWk = (const float*)d_in[6];   const float* sbk = (const float*)d_in[7];
  const float* sWv = (const float*)d_in[8];   const float* sbv = (const float*)d_in[9];
  const float* sWo = (const float*)d_in[10];  const float* sbo = (const float*)d_in[11];
  const float* cWq = (const float*)d_in[12];  const float* cbq = (const float*)d_in[13];
  const float* cWk = (const float*)d_in[14];  const float* cbk = (const float*)d_in[15];
  const float* cWv = (const float*)d_in[16];  const float* cbv = (const float*)d_in[17];
  const float* cWo = (const float*)d_in[18];  const float* cbo = (const float*)d_in[19];
  const float* W1  = (const float*)d_in[20];  const float* b1  = (const float*)d_in[21];
  const float* W2  = (const float*)d_in[22];  const float* b2  = (const float*)d_in[23];

  const size_t WDD = (size_t)D_SZ * D_SZ;          // 1M
  const size_t WDF = (size_t)D_SZ * DFF_SZ;        // 4M
  const size_t TOK = (size_t)B_SZ * L_SZ;          // 4096 rows
  const size_t XE  = TOK * D_SZ;                   // 4M
  const size_t HE  = TOK * DFF_SZ;                 // 16M

  char* ws = (char*)d_ws;
  size_t off = 0;
  auto take = [&](size_t bytes) { char* p = ws + off; off += (bytes + 255) & ~(size_t)255; return p; };

  uint16_t* wqs = (uint16_t*)take(WDD * 2);
  uint16_t* wks = (uint16_t*)take(WDD * 2);
  uint16_t* wvs = (uint16_t*)take(WDD * 2);
  uint16_t* wos = (uint16_t*)take(WDD * 2);
  uint16_t* wqc = (uint16_t*)take(WDD * 2);
  uint16_t* wkc = (uint16_t*)take(WDD * 2);
  uint16_t* wvc = (uint16_t*)take(WDD * 2);
  uint16_t* woc = (uint16_t*)take(WDD * 2);
  uint16_t* w1b = (uint16_t*)take(WDF * 2);
  uint16_t* w2b = (uint16_t*)take(WDF * 2);
  uint16_t* x0b = (uint16_t*)take(XE * 2);
  uint16_t* enb = (uint16_t*)take(XE * 2);
  uint16_t* qb  = (uint16_t*)take(XE * 2);
  uint16_t* kb  = (uint16_t*)take(XE * 2);
  uint16_t* vb  = (uint16_t*)take(XE * 2);
  uint16_t* ab  = (uint16_t*)take(XE * 2);
  uint16_t* x1b = (uint16_t*)take(XE * 2);
  uint16_t* x2b = (uint16_t*)take(XE * 2);
  uint16_t* hb  = (uint16_t*)take(HE * 2);
  float*    x1f = (float*)take(XE * 4);
  float*    x2f = (float*)take(XE * 4);
  float*    out = (float*)d_out;

  dim3 blk(256, 1, 1);
  auto cast = [&](const float* s, uint16_t* d, size_t n) {
    cast_f32_bf16<<<dim3((unsigned)((n + 255) / 256)), blk, 0, stream>>>(s, d, (int)n);
  };

  // --- demote to bf16 ---
  cast(tgt, x0b, XE);   cast(enc, enb, XE);
  cast(sWq, wqs, WDD);  cast(sWk, wks, WDD); cast(sWv, wvs, WDD); cast(sWo, wos, WDD);
  cast(cWq, wqc, WDD);  cast(cWk, wkc, WDD); cast(cWv, wvc, WDD); cast(cWo, woc, WDD);
  cast(W1,  w1b, WDF);  cast(W2,  w2b, WDF);

  const dim3 gDD((unsigned)(D_SZ / 64),   (unsigned)(TOK / 256));
  const dim3 gDF((unsigned)(DFF_SZ / 64), (unsigned)(TOK / 256));
  const dim3 gAtt((unsigned)(L_SZ / 128), (unsigned)H_SZ, (unsigned)B_SZ);

  // --- self attention ---
  gemm_bf16<false,false,false,true><<<gDD, blk, 0, stream>>>(x0b, wqs, sbq, nullptr, nullptr, qb, (int)TOK, D_SZ, D_SZ);
  gemm_bf16<false,false,false,true><<<gDD, blk, 0, stream>>>(x0b, wks, sbk, nullptr, nullptr, kb, (int)TOK, D_SZ, D_SZ);
  gemm_bf16<false,false,false,true><<<gDD, blk, 0, stream>>>(x0b, wvs, sbv, nullptr, nullptr, vb, (int)TOK, D_SZ, D_SZ);
  flash_attn_bf16<true><<<gAtt, blk, 0, stream>>>(qb, kb, vb, ab);
  gemm_bf16<false,true,true,true><<<gDD, blk, 0, stream>>>(ab, wos, sbo, tgt, x1f, x1b, (int)TOK, D_SZ, D_SZ);

  // --- cross attention ---
  gemm_bf16<false,false,false,true><<<gDD, blk, 0, stream>>>(x1b, wqc, cbq, nullptr, nullptr, qb, (int)TOK, D_SZ, D_SZ);
  gemm_bf16<false,false,false,true><<<gDD, blk, 0, stream>>>(enb, wkc, cbk, nullptr, nullptr, kb, (int)TOK, D_SZ, D_SZ);
  gemm_bf16<false,false,false,true><<<gDD, blk, 0, stream>>>(enb, wvc, cbv, nullptr, nullptr, vb, (int)TOK, D_SZ, D_SZ);
  flash_attn_bf16<false><<<gAtt, blk, 0, stream>>>(qb, kb, vb, ab);
  gemm_bf16<false,true,true,true><<<gDD, blk, 0, stream>>>(ab, woc, cbo, x1f, x2f, x2b, (int)TOK, D_SZ, D_SZ);

  // --- FFN ---
  gemm_bf16<true,false,false,true><<<gDF, blk, 0, stream>>>(x2b, w1b, b1, nullptr, nullptr, hb, (int)TOK, DFF_SZ, D_SZ);
  gemm_bf16<false,true,true,false><<<gDD, blk, 0, stream>>>(hb, w2b, b2, x2f, out, nullptr, (int)TOK, D_SZ, DFF_SZ);
}
